// Experts_18193481466108
// MI455X (gfx1250) — compile-verified, matrix operand order checked by reading, not run on hardware
//
#include <hip/hip_runtime.h>
#include <hip/hip_bf16.h>

typedef __attribute__((ext_vector_type(16))) __bf16 v16bf;
typedef __attribute__((ext_vector_type(8)))  __bf16 v8bf;
typedef __attribute__((ext_vector_type(4)))  __bf16 v4bf;
typedef __attribute__((ext_vector_type(8)))  float  v8f;
typedef __attribute__((ext_vector_type(4)))  float  v4f;

#define NE 8
#define NC 2048
#define ND 1024
#define NF 4096

#define TM 128
#define TN 128
#define TK 32
#define LDT 40   // padded LDS row stride (elements); 80B rows -> 16B aligned, conflict-free

#define AS1 __attribute__((address_space(1)))
#define AS3 __attribute__((address_space(3)))

typedef int v4i_ld __attribute__((vector_size(16)));   // matches builtin param type

#if defined(__has_builtin)
#if __has_builtin(__builtin_amdgcn_global_load_async_to_lds_b128)
#define HAVE_ASYNC_LDS 1
#endif
#endif

__device__ __forceinline__ void wait_async0() {
#if defined(HAVE_ASYNC_LDS)
#if __has_builtin(__builtin_amdgcn_s_wait_asynccnt)
    __builtin_amdgcn_s_wait_asynccnt(0);
#else
    asm volatile("s_wait_asynccnt 0x0" ::: "memory");
#endif
#endif
}

__device__ __forceinline__ void async_copy16(const void* g, void* l) {
#if defined(HAVE_ASYNC_LDS)
    __builtin_amdgcn_global_load_async_to_lds_b128(
        (AS1 v4i_ld*)g, (AS3 v4i_ld*)l, 0, 0);
#endif
}

// gelu(x) = 0.5x(1+tanh(z)) == x * sigmoid(2z),  z = sqrt(2/pi)(x + 0.044715 x^3)
__device__ __forceinline__ float gelu_fast(float x) {
    float z2 = x * (1.5957691216f + 0.0713548163f * x * x);   // 2*z
    float s  = __frcp_rn(1.0f + __expf(-z2));                 // v_exp_f32 + v_rcp_f32
    return x * s;
}

// Load one 16x32 bf16 WMMA fragment from LDS laid out [row][k], row stride LDT.
// Lane l (l<16): row l, K in {khalf*8+0..7, 16+khalf*8+0..7} per ISA A-layout.
__device__ __forceinline__ v16bf load_frag(const __bf16* lds, int row, int khalf) {
    const __bf16* p = lds + row * LDT + khalf * 8;
    v8bf lo = *(const v8bf*)(p);
    v8bf hi = *(const v8bf*)(p + 16);
    v16bf r;
#pragma unroll
    for (int i = 0; i < 8; ++i) { r[i] = lo[i]; r[i + 8] = hi[i]; }
    return r;
}

// ---------------- Kernel 1: H = gelu(X @ W1 + b1), H stored bf16 ----------------
__global__ __launch_bounds__(256)
void ffn1_kernel(const float* __restrict__ X, const float* __restrict__ W1,
                 const float* __restrict__ B1, __bf16* __restrict__ H) {
    __shared__ __bf16 Alds[2][TM * LDT];
    __shared__ __bf16 Blds[2][TN * LDT];

    const int e  = blockIdx.z;
    const int m0 = blockIdx.y * TM;
    const int n0 = blockIdx.x * TN;

    const float* Xe  = X  + (size_t)e * NC * ND;
    const float* W1e = W1 + (size_t)e * ND * NF;

    const int tid   = threadIdx.x;
    const int lane  = tid & 31;
    const int wave  = tid >> 5;
    const int waveM = (wave & 3) * 32;
    const int waveN = (wave >> 2) * 64;
    const int lrow  = lane & 15;
    const int khalf = lane >> 4;

    v8f acc[2][4];
#pragma unroll
    for (int i = 0; i < 2; ++i)
#pragma unroll
        for (int j = 0; j < 4; ++j)
#pragma unroll
            for (int r = 0; r < 8; ++r) acc[i][j][r] = 0.0f;

    v4f areg[4], breg[4];

    // ---- prologue: stage tile k=0 into registers (8 batched b128 loads) ----
#pragma unroll
    for (int i = 0; i < 4; ++i) {
        int q = tid + 256 * i, row = q >> 3, c4 = q & 7;
        areg[i] = *(const v4f*)(Xe + (size_t)(m0 + row) * ND + c4 * 4);
    }
#pragma unroll
    for (int i = 0; i < 4; ++i) {
        int q = tid + 256 * i, kr = q >> 5, c4 = q & 31;
        breg[i] = *(const v4f*)(W1e + (size_t)kr * NF + n0 + c4 * 4);
    }

    const int nk = ND / TK;
    for (int t = 0; t < nk; ++t) {
        const int buf = t & 1;
        // ---- convert staged registers and commit to LDS[buf] ----
#pragma unroll
        for (int i = 0; i < 4; ++i) {
            int q = tid + 256 * i, row = q >> 3, c4 = q & 7;
            v4bf b;
#pragma unroll
            for (int j = 0; j < 4; ++j) b[j] = (__bf16)areg[i][j];
            *(v4bf*)&Alds[buf][row * LDT + c4 * 4] = b;
        }
#pragma unroll
        for (int i = 0; i < 4; ++i) {
            int q = tid + 256 * i, kr = q >> 5, c4 = q & 31;
#pragma unroll
            for (int j = 0; j < 4; ++j)
                Blds[buf][(c4 * 4 + j) * LDT + kr] = (__bf16)breg[i][j];
        }
        __syncthreads();

        // ---- issue next tile's global loads (overlap with WMMA below) ----
        if (t + 1 < nk) {
            const int k0 = (t + 1) * TK;
#pragma unroll
            for (int i = 0; i < 4; ++i) {
                int q = tid + 256 * i, row = q >> 3, c4 = q & 7;
                areg[i] = *(const v4f*)(Xe + (size_t)(m0 + row) * ND + k0 + c4 * 4);
            }
#pragma unroll
            for (int i = 0; i < 4; ++i) {
                int q = tid + 256 * i, kr = q >> 5, c4 = q & 31;
                breg[i] = *(const v4f*)(W1e + (size_t)(k0 + kr) * NF + n0 + c4 * 4);
            }
            if (t + 2 < nk) {
                __builtin_prefetch(Xe  + (size_t)(m0 + (tid >> 1)) * ND + k0 + TK, 0, 0);
                __builtin_prefetch(W1e + (size_t)(k0 + TK + (tid >> 3)) * NF + n0, 0, 0);
            }
        }

        // ---- consume LDS[buf] ----
        v16bf af[2], bfr[4];
#pragma unroll
        for (int i = 0; i < 2; ++i) af[i]  = load_frag(Alds[buf], waveM + i * 16 + lrow, khalf);
#pragma unroll
        for (int j = 0; j < 4; ++j) bfr[j] = load_frag(Blds[buf], waveN + j * 16 + lrow, khalf);
#pragma unroll
        for (int i = 0; i < 2; ++i)
#pragma unroll
            for (int j = 0; j < 4; ++j)
                acc[i][j] = __builtin_amdgcn_wmma_f32_16x16x32_bf16(
                    false, af[i], false, bfr[j], (short)0, acc[i][j], false, false);
        // double buffer: single barrier per iteration is sufficient
    }

    // ---- epilogue: bias + gelu, store bf16 H ----
    const int rofs = khalf * 8;
#pragma unroll
    for (int i = 0; i < 2; ++i)
#pragma unroll
        for (int j = 0; j < 4; ++j) {
            int col = n0 + waveN + j * 16 + lrow;
            float bias = B1[(size_t)e * NF + col];
#pragma unroll
            for (int r = 0; r < 8; ++r) {
                int row = m0 + waveM + i * 16 + rofs + r;
                float v = gelu_fast(acc[i][j][r] + bias);
                H[(size_t)e * NC * NF + (size_t)row * NF + col] = (__bf16)v;
            }
        }
}

// ---------------- Kernel 2: out = H @ W2 + b2 (f32 out) ----------------
__global__ __launch_bounds__(256)
void ffn2_kernel(const __bf16* __restrict__ H, const float* __restrict__ W2,
                 const float* __restrict__ B2, float* __restrict__ Out) {
    __shared__ __bf16 Alds[2][TM * LDT];
    __shared__ __bf16 Blds[2][TN * LDT];

    const int e  = blockIdx.z;
    const int m0 = blockIdx.y * TM;
    const int n0 = blockIdx.x * TN;

    const __bf16* He  = H  + (size_t)e * NC * NF;
    const float*  W2e = W2 + (size_t)e * NF * ND;

    const int tid   = threadIdx.x;
    const int lane  = tid & 31;
    const int wave  = tid >> 5;
    const int waveM = (wave & 3) * 32;
    const int waveN = (wave >> 2) * 64;
    const int lrow  = lane & 15;
    const int khalf = lane >> 4;

    v8f acc[2][4];
#pragma unroll
    for (int i = 0; i < 2; ++i)
#pragma unroll
        for (int j = 0; j < 4; ++j)
#pragma unroll
            for (int r = 0; r < 8; ++r) acc[i][j][r] = 0.0f;

    v4f breg[4];
#if !defined(HAVE_ASYNC_LDS)
    v8bf areg[2];
#endif

    // ---- prologue: stage tile k=0 ----
#if defined(HAVE_ASYNC_LDS)
#pragma unroll
    for (int i = 0; i < 2; ++i) {
        int q = tid + 256 * i, row = q >> 2, c8 = q & 3;
        async_copy16(He + (size_t)(m0 + row) * NF + c8 * 8,
                     &Alds[0][row * LDT + c8 * 8]);
    }
#else
#pragma unroll
    for (int i = 0; i < 2; ++i) {
        int q = tid + 256 * i, row = q >> 2, c8 = q & 3;
        areg[i] = *(const v8bf*)(He + (size_t)(m0 + row) * NF + c8 * 8);
    }
#endif
#pragma unroll
    for (int i = 0; i < 4; ++i) {
        int q = tid + 256 * i, kr = q >> 5, c4 = q & 31;
        breg[i] = *(const v4f*)(W2e + (size_t)kr * ND + n0 + c4 * 4);
    }

    const int nk = NF / TK;
    for (int t = 0; t < nk; ++t) {
        const int buf = t & 1;
        // ---- commit staged B to LDS[buf]; A arrives via async (or regs) ----
#pragma unroll
        for (int i = 0; i < 4; ++i) {
            int q = tid + 256 * i, kr = q >> 5, c4 = q & 31;
#pragma unroll
            for (int j = 0; j < 4; ++j)
                Blds[buf][(c4 * 4 + j) * LDT + kr] = (__bf16)breg[i][j];
        }
#if defined(HAVE_ASYNC_LDS)
        wait_async0();     // A tile for LDS[buf] landed
#else
#pragma unroll
        for (int i = 0; i < 2; ++i) {
            int q = tid + 256 * i, row = q >> 2, c8 = q & 3;
            *(v8bf*)&Alds[buf][row * LDT + c8 * 8] = areg[i];
        }
#endif
        __syncthreads();

        // ---- issue next tile's loads (overlap with WMMA below) ----
        if (t + 1 < nk) {
            const int k0 = (t + 1) * TK;
#if defined(HAVE_ASYNC_LDS)
#pragma unroll
            for (int i = 0; i < 2; ++i) {
                int q = tid + 256 * i, row = q >> 2, c8 = q & 3;
                async_copy16(He + (size_t)(m0 + row) * NF + k0 + c8 * 8,
                             &Alds[buf ^ 1][row * LDT + c8 * 8]);
            }
#else
#pragma unroll
            for (int i = 0; i < 2; ++i) {
                int q = tid + 256 * i, row = q >> 2, c8 = q & 3;
                areg[i] = *(const v8bf*)(He + (size_t)(m0 + row) * NF + k0 + c8 * 8);
            }
#endif
#pragma unroll
            for (int i = 0; i < 4; ++i) {
                int q = tid + 256 * i, kr = q >> 5, c4 = q & 31;
                breg[i] = *(const v4f*)(W2e + (size_t)(k0 + kr) * ND + n0 + c4 * 4);
            }
            if (t + 2 < nk) {
                __builtin_prefetch(He  + (size_t)(m0 + (tid >> 1)) * NF + k0 + TK, 0, 0);
                __builtin_prefetch(W2e + (size_t)(k0 + TK + (tid >> 3)) * ND + n0, 0, 0);
            }
        }

        // ---- consume LDS[buf] ----
        v16bf af[2], bfr[4];
#pragma unroll
        for (int i = 0; i < 2; ++i) af[i]  = load_frag(Alds[buf], waveM + i * 16 + lrow, khalf);
#pragma unroll
        for (int j = 0; j < 4; ++j) bfr[j] = load_frag(Blds[buf], waveN + j * 16 + lrow, khalf);
#pragma unroll
        for (int i = 0; i < 2; ++i)
#pragma unroll
            for (int j = 0; j < 4; ++j)
                acc[i][j] = __builtin_amdgcn_wmma_f32_16x16x32_bf16(
                    false, af[i], false, bfr[j], (short)0, acc[i][j], false, false);
    }

    const int rofs = khalf * 8;
#pragma unroll
    for (int i = 0; i < 2; ++i)
#pragma unroll
        for (int j = 0; j < 4; ++j) {
            int col = n0 + waveN + j * 16 + lrow;
            float bias = B2[(size_t)e * ND + col];
#pragma unroll
            for (int r = 0; r < 8; ++r) {
                int row = m0 + waveM + i * 16 + rofs + r;
                Out[(size_t)e * NC * ND + (size_t)row * ND + col] = acc[i][j][r] + bias;
            }
        }
}

extern "C" void kernel_launch(void* const* d_in, const int* in_sizes, int n_in,
                              void* d_out, int out_size, void* d_ws, size_t ws_size,
                              hipStream_t stream) {
    // setup_inputs order: inputs, w1, b1, w2, b2 (all fp32)
    const float* X  = (const float*)d_in[0];   // [1, E, C, D]
    const float* W1 = (const float*)d_in[1];   // [E, D, F]
    const float* B1 = (const float*)d_in[2];   // [E, 1, F]
    const float* W2 = (const float*)d_in[3];   // [E, F, D]
    const float* B2 = (const float*)d_in[4];   // [E, 1, D]
    float* out = (float*)d_out;                // [1, E, C, D]

    __bf16* H = (__bf16*)d_ws;                 // [E, C, F] bf16 intermediate (128 MB)

    dim3 blk(256, 1, 1);
    dim3 g1(NF / TN, NC / TM, NE);             // 32 x 16 x 8
    ffn1_kernel<<<g1, blk, 0, stream>>>(X, W1, B1, H);
    dim3 g2(ND / TN, NC / TM, NE);             // 8 x 16 x 8
    ffn2_kernel<<<g2, blk, 0, stream>>>(H, W2, B2, out);
}
